// CrossAttention_67903432950172
// MI455X (gfx1250) — matrix operand order repaired and checked by reading, not compiled, not run on hardware
//
#include <hip/hip_runtime.h>
#include <hip/hip_bf16.h>

// ---------------------------------------------------------------------------
// CrossAttention on MI455X (gfx1250, wave32, WMMA + TDM).
//
// - f16 WMMA (v_wmma_f32_16x16x32_f16) with fp32 accumulate: workload is
//   memory-bound (att_mean RMW ~1GB @ 23.3 TB/s ~= 45us floor), f16 matrix
//   ops keep the ~60 GFLOP of compute far below that floor; softmax is exact
//   fp32.
// - 16x2048 fp32 score strip (128KB) lives in LDS (possible only with CDNA5's
//   320KB/WGP) so per-head attention never spills to HBM.
// - P@V's B operand (V read column-tile-wise, 1KB stride) is staged through
//   LDS with the Tensor Data Mover (tensor_load_to_lds, TENSORcnt), double
//   buffered so the DMA overlaps the WMMA stream.
// ---------------------------------------------------------------------------

typedef __attribute__((ext_vector_type(16))) _Float16 v16h;
typedef __attribute__((ext_vector_type(8)))  float    v8f;
typedef __attribute__((ext_vector_type(4)))  unsigned v4u;
typedef __attribute__((ext_vector_type(8)))  unsigned v8u;

#define N_EMBD   512
#define N_HEAD   8
#define HEAD_DIM 64
#define B_       8
#define TX       1024
#define TC       2048

// ---- WMMA wrapper ----------------------------------------------------------
__device__ __forceinline__ v8f wmma_f16(v16h a, v16h b, v8f c) {
  // (neg_a, A, neg_b, B, c_mod, C, reuse_a, reuse_b)
  return __builtin_amdgcn_wmma_f32_16x16x32_f16(false, a, false, b, (short)0, c,
                                                false, false);
}

// ---- Fragment packing (wave32, contiguous k = 16*half + s, same for A and B) ---------------------------------
// A 16x32 f16: lane L -> row m = L&15; half = L>>4 selects K-group.
//   element 2v -> K = half*16 + 2*v; element 2v+1 -> K+1
__device__ __forceinline__ v16h pack_a_f16(const _Float16* __restrict__ src,
                                           int ld, int m, int half, int kbase) {
  v16h a;
#pragma unroll
  for (int v = 0; v < 8; ++v) {
    int k0 = half * 16 + 2 * v;
    a[2 * v]     = src[(size_t)m * ld + kbase + k0];
    a[2 * v + 1] = src[(size_t)m * ld + kbase + k0 + 1];
  }
  return a;
}

__device__ __forceinline__ v16h pack_a_f32(const float* __restrict__ src,
                                           int ld, int m, int half, int kbase) {
  v16h a;
#pragma unroll
  for (int v = 0; v < 8; ++v) {
    int k0 = half * 16 + 2 * v;
    a[2 * v]     = (_Float16)src[(size_t)m * ld + kbase + k0];
    a[2 * v + 1] = (_Float16)src[(size_t)m * ld + kbase + k0 + 1];
  }
  return a;
}

// B 32x16 f16: lane L -> col n = L&15; half = L>>4; element 2v+t -> K=16*half+2v+t
// pack_bT_*: source row-major [N,K] (Linear weight / K-matrix case)
__device__ __forceinline__ v16h pack_bT_f16(const _Float16* __restrict__ src,
                                            int ld, int n, int half, int kbase) {
  v16h b;
#pragma unroll
  for (int v = 0; v < 8; ++v)
#pragma unroll
    for (int t = 0; t < 2; ++t) {
      int k = half * 16 + 2 * v + t;
      b[2 * v + t] = src[(size_t)n * ld + kbase + k];
    }
  return b;
}

__device__ __forceinline__ v16h pack_bT_f32(const float* __restrict__ src,
                                            int ld, int n, int half, int kbase) {
  v16h b;
#pragma unroll
  for (int v = 0; v < 8; ++v)
#pragma unroll
    for (int t = 0; t < 2; ++t) {
      int k = half * 16 + 2 * v + t;
      b[2 * v + t] = (_Float16)src[(size_t)n * ld + kbase + k];
    }
  return b;
}

// B from an LDS-staged [32][64] f16 slab (row-major, rows = K dim)
__device__ __forceinline__ v16h pack_b_lds(const _Float16* slab, int col,
                                           int half) {
  v16h b;
#pragma unroll
  for (int v = 0; v < 8; ++v)
#pragma unroll
    for (int t = 0; t < 2; ++t) {
      int k = half * 16 + 2 * v + t;
      b[2 * v + t] = slab[k * HEAD_DIM + col];
    }
  return b;
}

// ---- Tensor Data Mover: DMA a 32-row x 64-col f16 slab (row stride 512) ----
// D# group0/group1 per CDNA5 ISA (2D tensor,
// VADDR2/3 = NULL -> 2-group form). Issued by one wave; completion via
// TENSORcnt then published with a workgroup barrier.
__device__ __forceinline__ void tdm_load_v_slab(const _Float16* gsrc,
                                                unsigned lds_byte_addr) {
  unsigned long long ga = (unsigned long long)(size_t)(const void*)gsrc;
  v4u g0;
  g0[0] = 1u;                                   // count=1, user descriptor
  g0[1] = lds_byte_addr;                        // lds_addr
  g0[2] = (unsigned)ga;                         // global_addr[31:0]
  g0[3] = (unsigned)((ga >> 32) & 0x1FFFFFFull) // global_addr[56:32]
          | (2u << 30);                         // type=2 ("image")
  v8u g1;
  g1[0] = 0x00010000u;      // wg_mask=0, data_size=1 (2B), no pad/iterate
  g1[1] = 64u << 16;        // tensor_dim0 = 64
  g1[2] = 32u << 16;        // tensor_dim0 hi=0, tensor_dim1 = 32
  g1[3] = 64u << 16;        // tensor_dim1 hi=0, tile_dim0 = 64
  g1[4] = 32u;              // tile_dim1 = 32, tile_dim2 = 0
  g1[5] = (unsigned)N_EMBD; // tensor_dim0_stride = 512 elements
  g1[6] = 0u;               // stride hi / dim1_stride lo
  g1[7] = 0u;               // dim1_stride hi
  asm volatile("tensor_load_to_lds %0, %1" ::"s"(g0), "s"(g1) : "memory");
}

// ---- Kernel 1: out_f16[M,512] = in_f32[M,512] @ W^T + b --------------------
// grid = (N/64, M/16), block = 32: one wave owns a 16x64 strip (4 WMMA tiles,
// one A-pack amortized over 4 B-packs -> 4x less A fetch traffic).
__global__ void proj_f32_f16(const float* __restrict__ in,
                             const float* __restrict__ W,
                             const float* __restrict__ bias,
                             _Float16* __restrict__ out) {
  const int K = N_EMBD, N = N_EMBD;
  const int nt0 = blockIdx.x * 4, mt = blockIdx.y;
  const int lane = threadIdx.x & 31;
  const int m = lane & 15, half = lane >> 4, n = lane & 15;

  const float* arow = in + (size_t)(mt * 16) * K;
  v8f acc[4] = {};
  for (int kk = 0; kk < K; kk += 32) {
    v16h a = pack_a_f32(arow, K, m, half, kk);
#pragma unroll
    for (int j = 0; j < 4; ++j) {
      v16h bm = pack_bT_f32(W + (size_t)((nt0 + j) * 16) * K, K, n, half, kk);
      acc[j] = wmma_f16(a, bm, acc[j]);
    }
  }
#pragma unroll
  for (int j = 0; j < 4; ++j) {
    const float bb = bias[(nt0 + j) * 16 + n];
#pragma unroll
    for (int r = 0; r < 8; ++r) {
      int row = mt * 16 + 8 * half + r;
      out[(size_t)row * N + (nt0 + j) * 16 + n] = (_Float16)(acc[j][r] + bb);
    }
  }
}

// ---- Kernel 3: out_f32[M,512] = in_f16[M,512] @ W^T + b --------------------
__global__ void proj_f16_f32(const _Float16* __restrict__ in,
                             const float* __restrict__ W,
                             const float* __restrict__ bias,
                             float* __restrict__ out) {
  const int K = N_EMBD, N = N_EMBD;
  const int nt0 = blockIdx.x * 4, mt = blockIdx.y;
  const int lane = threadIdx.x & 31;
  const int m = lane & 15, half = lane >> 4, n = lane & 15;

  const _Float16* arow = in + (size_t)(mt * 16) * K;
  v8f acc[4] = {};
  for (int kk = 0; kk < K; kk += 32) {
    v16h a = pack_a_f16(arow, K, m, half, kk);
#pragma unroll
    for (int j = 0; j < 4; ++j) {
      v16h bm = pack_bT_f32(W + (size_t)((nt0 + j) * 16) * K, K, n, half, kk);
      acc[j] = wmma_f16(a, bm, acc[j]);
    }
  }
#pragma unroll
  for (int j = 0; j < 4; ++j) {
    const float bb = bias[(nt0 + j) * 16 + n];
#pragma unroll
    for (int r = 0; r < 8; ++r) {
      int row = mt * 16 + 8 * half + r;
      out[(size_t)row * N + (nt0 + j) * 16 + n] = acc[j][r] + bb;
    }
  }
}

// ---- Kernel 2: fused attention ---------------------------------------------
// grid = (TX/16, B), block = 256 (8 waves). One workgroup owns a 16-row query
// tile of one batch and loops over all 8 heads -> att_mean accumulation is an
// exclusive RMW. LDS: S[16][2048] fp32 (128KB) + 2 x V-slab (32x64 f16, 8KB).
__global__ void attn_kernel(const _Float16* __restrict__ qh,
                            const _Float16* __restrict__ kh,
                            const _Float16* __restrict__ vh,
                            _Float16* __restrict__ yh,
                            float* __restrict__ attm) {
  extern __shared__ float S[];  // [16][TC] then V double buffer
  _Float16* vslab0 = (_Float16*)(S + 16 * TC);
  _Float16* vslab1 = vslab0 + 32 * HEAD_DIM;

  const int qt = blockIdx.x;
  const int b  = blockIdx.y;
  const int tid  = threadIdx.x;
  const int wid  = tid >> 5;
  const int lane = tid & 31;
  const int m = lane & 15, half = lane >> 4, n = lane & 15;
  const int q0 = qt * 16;

  for (int h = 0; h < N_HEAD; ++h) {
    // ---- S = (Q . K^T) * 1/sqrt(hd), 128 col-tiles split across 8 waves ----
    const _Float16* qrow = qh + ((size_t)(b * TX + q0)) * N_EMBD + h * HEAD_DIM;
    v16h a0 = pack_a_f16(qrow, N_EMBD, m, half, 0);
    v16h a1 = pack_a_f16(qrow, N_EMBD, m, half, 32);
    for (int ct = wid; ct < TC / 16; ct += 8) {
      const _Float16* krow =
          kh + ((size_t)(b * TC + ct * 16)) * N_EMBD + h * HEAD_DIM;
      v8f acc = {};
      v16h b0 = pack_bT_f16(krow, N_EMBD, n, half, 0);
      acc = wmma_f16(a0, b0, acc);
      v16h b1 = pack_bT_f16(krow, N_EMBD, n, half, 32);
      acc = wmma_f16(a1, b1, acc);
#pragma unroll
      for (int r = 0; r < 8; ++r)
        S[(size_t)(8 * half + r) * TC + ct * 16 + n] = acc[r] * 0.125f;
    }
    __syncthreads();

    // ---- exact fp32 softmax per row; fold P/8 into att_mean ----
    {
      const int row = tid >> 4;  // 16 threads per row
      const int sub = tid & 15;
      float* srow = S + (size_t)row * TC;
      float mx = -3.4e38f;
      for (int c = sub; c < TC; c += 16) mx = fmaxf(mx, srow[c]);
#pragma unroll
      for (int off = 8; off >= 1; off >>= 1)
        mx = fmaxf(mx, __shfl_xor(mx, off, 16));
      float sum = 0.f;
      for (int c = sub; c < TC; c += 16) {
        float e = __expf(srow[c] - mx);
        srow[c] = e;
        sum += e;
      }
#pragma unroll
      for (int off = 8; off >= 1; off >>= 1) sum += __shfl_xor(sum, off, 16);
      const float inv = 1.0f / sum;
      float* arow_out = attm + ((size_t)(b * TX + q0 + row)) * TC;
      for (int c = sub; c < TC; c += 16) {
        float p = srow[c] * inv;
        srow[c] = p;
        float prev = (h == 0) ? 0.f : arow_out[c];
        arow_out[c] = prev + p * (1.0f / N_HEAD);
      }
    }
    __syncthreads();

    // ---- Y_h = P @ V_h, V staged via TDM (double buffered) ----
    const _Float16* vbase = vh + ((size_t)(b * TC)) * N_EMBD + h * HEAD_DIM;
    if (wid == 0) {
      tdm_load_v_slab(vbase, (unsigned)(unsigned long long)(void*)vslab0);
      __builtin_amdgcn_s_wait_tensorcnt(0);
    }
    __syncthreads();

    v8f acc = {};
    int cur = 0;
    for (int kk = 0; kk < TC; kk += 32) {
      if (wid == 0 && (kk + 32) < TC)
        tdm_load_v_slab(vbase + (size_t)(kk + 32) * N_EMBD,
                        (unsigned)(unsigned long long)(void*)(cur ? vslab0
                                                                  : vslab1));
      if (wid < 4) {
        v16h a  = pack_a_f32(S, TC, m, half, kk);
        v16h bm = pack_b_lds(cur ? vslab1 : vslab0, wid * 16 + n, half);
        acc = wmma_f16(a, bm, acc);
      }
      if (wid == 0) __builtin_amdgcn_s_wait_tensorcnt(0);
      __syncthreads();
      cur ^= 1;
    }
    if (wid < 4) {
#pragma unroll
      for (int r = 0; r < 8; ++r) {
        int row = q0 + 8 * half + r;
        yh[((size_t)(b * TX + row)) * N_EMBD + h * HEAD_DIM + wid * 16 + n] =
            (_Float16)acc[r];
      }
    }
    __syncthreads();
  }
}

// ---------------------------------------------------------------------------
extern "C" void kernel_launch(void* const* d_in, const int* in_sizes, int n_in,
                              void* d_out, int out_size, void* d_ws,
                              size_t ws_size, hipStream_t stream) {
  (void)in_sizes; (void)n_in; (void)out_size; (void)ws_size;
  const float* x   = (const float*)d_in[0];
  const float* ctx = (const float*)d_in[1];
  const float* Wq  = (const float*)d_in[2];
  const float* bq  = (const float*)d_in[3];
  const float* Wk  = (const float*)d_in[4];
  const float* bk  = (const float*)d_in[5];
  const float* Wv  = (const float*)d_in[6];
  const float* bv  = (const float*)d_in[7];
  const float* Wo  = (const float*)d_in[8];
  const float* bo  = (const float*)d_in[9];

  float* y_out = (float*)d_out;                             // [B,TX,C]
  float* attm  = (float*)d_out + (size_t)B_ * TX * N_EMBD;  // [B,TX,TC]

  char* ws = (char*)d_ws;
  _Float16* qh = (_Float16*)ws;  ws += (size_t)B_ * TX * N_EMBD * sizeof(_Float16);
  _Float16* kh = (_Float16*)ws;  ws += (size_t)B_ * TC * N_EMBD * sizeof(_Float16);
  _Float16* vh = (_Float16*)ws;  ws += (size_t)B_ * TC * N_EMBD * sizeof(_Float16);
  _Float16* yh = (_Float16*)ws;

  const dim3 blk(32);
  proj_f32_f16<<<dim3(N_EMBD / 64, B_ * TX / 16), blk, 0, stream>>>(x, Wq, bq, qh);
  proj_f32_f16<<<dim3(N_EMBD / 64, B_ * TC / 16), blk, 0, stream>>>(ctx, Wk, bk, kh);
  proj_f32_f16<<<dim3(N_EMBD / 64, B_ * TC / 16), blk, 0, stream>>>(ctx, Wv, bv, vh);

  // 128KB score strip + 8KB V double-buffer (CDNA5: 320KB LDS per WGP)
  const size_t shmem = (size_t)16 * TC * sizeof(float) +
                       (size_t)2 * 32 * HEAD_DIM * sizeof(_Float16);
  attn_kernel<<<dim3(TX / 16, B_), dim3(256), shmem, stream>>>(qh, kh, vh, yh,
                                                               attm);

  proj_f16_f32<<<dim3(N_EMBD / 64, B_ * TX / 16), blk, 0, stream>>>(yh, Wo, bo,
                                                                    y_out);
}
